// EF_encoder_73177652789702
// MI455X (gfx1250) — matrix operand order repaired and checked by reading, not compiled, not run on hardware
//
#include <hip/hip_runtime.h>
#include <math.h>

// ---------------------------------------------------------------------------
// EF point-cloud encoder forward for gfx1250 (MI455X).
// Dense linear algebra via v_wmma_f32_16x16x32_f16 (f16 in, f32 accumulate).
// GEMM is templated on transpose flags; the strided operand is staged through
// an LDS tile (coalesced global fills -> float4 ds_load_b128 clauses), the
// contiguous operand is read directly from global (compiler merges to b128).
// Irregular ops (kNN top-16, FPS, gathers, fused DenseGCN) are VALU kernels.
// ---------------------------------------------------------------------------

typedef __attribute__((ext_vector_type(16))) _Float16 v16h;
typedef __attribute__((ext_vector_type(8)))  float    v8f;

#define NEG_BIG (-3.0e38f)
#define LB_STRIDE 36   // floats; 144B rows -> fragment bases 16B aligned

static inline unsigned cdiv(long a, long b) { return (unsigned)((a + b - 1) / b); }

// ---------------------------------------------------------------------------
// WMMA GEMM: C[M,N] = op(A)[M,K] * op(B)[K,N] (+bias[M]) (+rel) (ReLU)
// op(A): TA ? A[k*lda+m] : A[m*lda+k];  op(B): TB ? B[n*ldb+k] : B[k*ldb+n]
// rel epilogue: C += rowv[m] - colv[n] + rbias[0]
// One wave computes a 16(M) x 32(N) tile pair; grid.z = batch.
// Fragment layouts used here (contiguous k order, same for A and B):
//   A 16x32 f16: elem e -> K = half*16 + e, M = lane&15
//   B 32x16 f16: elem e -> K = half*16 + e, N = lane&15
//   D 16x16 f32: vgpr r -> M = r + 8*half, N = lane&15
// ---------------------------------------------------------------------------
template <int TA, int TB>
__launch_bounds__(32)
__global__ void k_gemm_t(const float* __restrict__ A, const float* __restrict__ B,
                         float* __restrict__ C, const float* __restrict__ bias,
                         int M, int N, int K, int lda, int ldb, int ldc,
                         long sA, long sB, long sC, int doRelu,
                         const float* __restrict__ rowv, const float* __restrict__ colv,
                         const float* __restrict__ rbias, long sRow, long sCol)
{
  __shared__ float lb[32 * LB_STRIDE];  // B tile, transposed: [n_local][k_local]
  __shared__ float la[16 * LB_STRIDE];  // A tile (TA only):   [m_local][k_local]

  const int bz = blockIdx.z;
  A += (long)bz * sA;  B += (long)bz * sB;  C += (long)bz * sC;
  const int n0   = blockIdx.x * 32;
  const int m0   = blockIdx.y * 16;
  const int lane = threadIdx.x;
  const int half = lane >> 4;
  const int l16  = lane & 15;

  const int am  = m0 + l16;
  const int bn0 = n0 + l16;
  const int bn1 = bn0 + 16;

  v8f acc0 = {}, acc1 = {};
  int k0 = 0;

  const bool interior = (m0 + 16 <= M) && (n0 + 32 <= N);
  if (interior) {
    for (; k0 + 32 <= K; k0 += 32) {
      // ---- stage strided operands in LDS (coalesced fills) ----
      if (!TB) {
        // row k = i, col = n0 + lane  ->  lb[lane][i]
        #pragma unroll
        for (int i = 0; i < 32; ++i)
          lb[lane * LB_STRIDE + i] = B[(long)(k0 + i) * ldb + n0 + lane];
      }
      if (TA) {
        // row k = 2i+half, col m0 + l16  ->  la[l16][2i+half]
        #pragma unroll
        for (int i = 0; i < 16; ++i)
          la[l16 * LB_STRIDE + (i << 1) + half] =
              A[(long)(k0 + (i << 1) + half) * lda + m0 + l16];
      }
      if (!TB || TA) __syncthreads();  // single wave: lowers to a DS wait

      // ---- A fragment ----
      v16h af;
      if (!TA) {
        const float* pa = A + (long)am * lda + k0 + (half << 4);
        #pragma unroll
        for (int e = 0; e < 8; ++e) af[e] = (_Float16)pa[e];
        #pragma unroll
        for (int e = 0; e < 8; ++e) af[8 + e] = (_Float16)pa[8 + e];
      } else {
        // LDS rows are 16B aligned: read as float4s (one DS clause)
        const float4* pa = (const float4*)(la + l16 * LB_STRIDE + (half << 4));
        float4 q0 = pa[0], q1 = pa[1], q2 = pa[2], q3 = pa[3]; // +0,+4,+8,+12
        af[0]=(_Float16)q0.x; af[1]=(_Float16)q0.y; af[2]=(_Float16)q0.z; af[3]=(_Float16)q0.w;
        af[4]=(_Float16)q1.x; af[5]=(_Float16)q1.y; af[6]=(_Float16)q1.z; af[7]=(_Float16)q1.w;
        af[8]=(_Float16)q2.x; af[9]=(_Float16)q2.y; af[10]=(_Float16)q2.z; af[11]=(_Float16)q2.w;
        af[12]=(_Float16)q3.x; af[13]=(_Float16)q3.y; af[14]=(_Float16)q3.z; af[15]=(_Float16)q3.w;
      }

      // ---- B fragments (two N tiles) ----
      v16h b0, b1;
      if (!TB) {
        const float4* r0 = (const float4*)(lb + l16 * LB_STRIDE + (half << 4));
        const float4* r1 = (const float4*)(lb + (l16 + 16) * LB_STRIDE + (half << 4));
        float4 p0 = r0[0], p1 = r0[1], p2 = r0[2], p3 = r0[3];
        float4 q0 = r1[0], q1 = r1[1], q2 = r1[2], q3 = r1[3];
        b0[0]=(_Float16)p0.x; b0[1]=(_Float16)p0.y; b0[2]=(_Float16)p0.z; b0[3]=(_Float16)p0.w;
        b0[4]=(_Float16)p1.x; b0[5]=(_Float16)p1.y; b0[6]=(_Float16)p1.z; b0[7]=(_Float16)p1.w;
        b0[8]=(_Float16)p2.x; b0[9]=(_Float16)p2.y; b0[10]=(_Float16)p2.z; b0[11]=(_Float16)p2.w;
        b0[12]=(_Float16)p3.x; b0[13]=(_Float16)p3.y; b0[14]=(_Float16)p3.z; b0[15]=(_Float16)p3.w;
        b1[0]=(_Float16)q0.x; b1[1]=(_Float16)q0.y; b1[2]=(_Float16)q0.z; b1[3]=(_Float16)q0.w;
        b1[4]=(_Float16)q1.x; b1[5]=(_Float16)q1.y; b1[6]=(_Float16)q1.z; b1[7]=(_Float16)q1.w;
        b1[8]=(_Float16)q2.x; b1[9]=(_Float16)q2.y; b1[10]=(_Float16)q2.z; b1[11]=(_Float16)q2.w;
        b1[12]=(_Float16)q3.x; b1[13]=(_Float16)q3.y; b1[14]=(_Float16)q3.z; b1[15]=(_Float16)q3.w;
      } else {
        const float* pb0 = B + (long)bn0 * ldb + k0 + (half << 4);
        const float* pb1 = B + (long)bn1 * ldb + k0 + (half << 4);
        __builtin_prefetch(&pb0[32]);
        #pragma unroll
        for (int e = 0; e < 16; ++e) { b0[e] = (_Float16)pb0[e]; b1[e] = (_Float16)pb1[e]; }
      }

      acc0 = __builtin_amdgcn_wmma_f32_16x16x32_f16(false, af, false, b0,
                                                    (short)0, acc0, false, false);
      acc1 = __builtin_amdgcn_wmma_f32_16x16x32_f16(false, af, false, b1,
                                                    (short)0, acc1, false, false);
      if (!TB || TA) __syncthreads();  // protect LDS tile before next fill
    }
  }
  // guarded path: edge tiles and/or K tail (direct global, masked)
  for (; k0 < K; k0 += 32) {
    v16h af, b0, b1;
    #pragma unroll
    for (int e = 0; e < 16; ++e) {
      int kk = k0 + (half << 4) + e;
      float x = 0.f;
      if (am < M && kk < K)
        x = TA ? A[(long)kk * lda + am] : A[(long)am * lda + kk];
      af[e] = (_Float16)x;
    }
    #pragma unroll
    for (int e = 0; e < 16; ++e) {
      int kk = k0 + (half << 4) + e;
      float x0 = 0.f, x1 = 0.f;
      if (kk < K) {
        if (bn0 < N) x0 = TB ? B[(long)bn0 * ldb + kk] : B[(long)kk * ldb + bn0];
        if (bn1 < N) x1 = TB ? B[(long)bn1 * ldb + kk] : B[(long)kk * ldb + bn1];
      }
      b0[e] = (_Float16)x0;
      b1[e] = (_Float16)x1;
    }
    acc0 = __builtin_amdgcn_wmma_f32_16x16x32_f16(false, af, false, b0,
                                                  (short)0, acc0, false, false);
    acc1 = __builtin_amdgcn_wmma_f32_16x16x32_f16(false, af, false, b1,
                                                  (short)0, acc1, false, false);
  }

  // epilogue (both N tiles)
  #pragma unroll
  for (int tile = 0; tile < 2; ++tile) {
    int cn = (tile == 0) ? bn0 : bn1;
    if (cn >= N) continue;
    v8f acc = (tile == 0) ? acc0 : acc1;
    #pragma unroll
    for (int r = 0; r < 8; ++r) {
      int cm = m0 + r + (half << 3);
      if (cm < M) {
        float v = acc[r];
        if (bias) v += bias[cm];
        if (rowv) v += rowv[(long)bz * sRow + cm] - colv[(long)bz * sCol + cn] + rbias[0];
        if (doRelu) v = fmaxf(v, 0.f);
        C[(long)cm * ldc + cn] = v;
      }
    }
  }
}

// ---------------------------------------------------------------------------
// Elementwise / support kernels
// ---------------------------------------------------------------------------
__global__ void k_pos_enc(const float* __restrict__ x, float* __restrict__ pe,
                          int N, int B)
{
  long t = (long)blockIdx.x * blockDim.x + threadIdx.x;
  if (t >= (long)B * N) return;
  int b = (int)(t / N), n = (int)(t % N);
  float v0 = x[((long)b * 3 + 0) * N + n];
  float v1 = x[((long)b * 3 + 1) * N + n];
  float v2 = x[((long)b * 3 + 2) * N + n];
  float* o = pe + (long)b * 21 * N;
  o[0 * N + n] = v0; o[1 * N + n] = v1; o[2 * N + n] = v2;
  int ch = 3;
  float fs[3] = {1.f, 2.f, 4.f};
  for (int fi = 0; fi < 3; ++fi) {
    float f = fs[fi];
    o[(ch + 0) * N + n] = sinf(v0 * f); o[(ch + 1) * N + n] = sinf(v1 * f); o[(ch + 2) * N + n] = sinf(v2 * f);
    ch += 3;
    o[(ch + 0) * N + n] = cosf(v0 * f); o[(ch + 1) * N + n] = cosf(v1 * f); o[(ch + 2) * N + n] = cosf(v2 * f);
    ch += 3;
  }
}

__global__ void k_copy(float* __restrict__ dst, int ldd, long sd,
                       const float* __restrict__ src, int lds_, long ss,
                       int C, int N, int B)
{
  long t = (long)blockIdx.x * blockDim.x + threadIdx.x;
  if (t >= (long)B * C * N) return;
  int n = (int)(t % N);
  long r = t / N;
  int c = (int)(r % C), b = (int)(r / C);
  dst[(long)b * sd + (long)c * ldd + n] = src[(long)b * ss + (long)c * lds_ + n];
}

__global__ void k_sub(float* __restrict__ dst, int ldd, long sd,
                      const float* __restrict__ a, int la, long sa,
                      const float* __restrict__ bb, int lb, long sb,
                      int C, int N, int B)
{
  long t = (long)blockIdx.x * blockDim.x + threadIdx.x;
  if (t >= (long)B * C * N) return;
  int n = (int)(t % N);
  long r = t / N;
  int c = (int)(r % C), b = (int)(r / C);
  dst[(long)b * sd + (long)c * ldd + n] =
      a[(long)b * sa + (long)c * la + n] - bb[(long)b * sb + (long)c * lb + n];
}

__global__ void k_bn_res(float* __restrict__ dst, int ldd, long sd,
                         const float* __restrict__ xc, int lxc, long sxc,
                         const float* __restrict__ tt, int lt, long st,
                         const float* __restrict__ gamma, const float* __restrict__ beta,
                         int C, int N, int B)
{
  long t = (long)blockIdx.x * blockDim.x + threadIdx.x;
  if (t >= (long)B * C * N) return;
  int n = (int)(t % N);
  long r = t / N;
  int c = (int)(r % C), b = (int)(r / C);
  float inv = rsqrtf(1.f + 1e-5f);
  float v = tt[(long)b * st + (long)c * lt + n] * inv * gamma[c] + beta[c];
  v = fmaxf(v, 0.f);
  dst[(long)b * sd + (long)c * ldd + n] = xc[(long)b * sxc + (long)c * lxc + n] + v;
}

__global__ void k_proj(const float* __restrict__ pe, const float* __restrict__ w,
                       float* __restrict__ proj, int N, int B)
{
  long t = (long)blockIdx.x * blockDim.x + threadIdx.x;
  if (t >= (long)B * N) return;
  int b = (int)(t / N), n = (int)(t % N);
  const float* p = pe + (long)b * 21 * N;
  float s = 0.f;
  for (int c = 0; c < 21; ++c) s += p[(long)c * N + n] * w[c];
  proj[t] = s;
}

__global__ void k_sqnorm(const float* __restrict__ x, int ld, long xs,
                         float* __restrict__ out, int C, int N, int B)
{
  long t = (long)blockIdx.x * blockDim.x + threadIdx.x;
  if (t >= (long)B * N) return;
  int b = (int)(t / N), n = (int)(t % N);
  const float* xb = x + (long)b * xs;
  float s = 0.f;
  for (int c = 0; c < C; ++c) { float v = xb[(long)c * ld + n]; s += v * v; }
  out[t] = s;
}

// ---------------------------------------------------------------------------
// Softmax over rows + column renorm (renorm_softmax)
// ---------------------------------------------------------------------------
__global__ void k_softmax_rows(float* __restrict__ att, int ldm, long satt, int M)
{
  int row = blockIdx.x, b = blockIdx.y;
  float* p = att + (long)b * satt + (long)row * ldm;
  __shared__ float red[256];
  int tid = threadIdx.x, T = blockDim.x;
  float mx = NEG_BIG;
  for (int j = tid; j < M; j += T) mx = fmaxf(mx, p[j]);
  red[tid] = mx; __syncthreads();
  for (int off = T >> 1; off; off >>= 1) {
    if (tid < off) red[tid] = fmaxf(red[tid], red[tid + off]);
    __syncthreads();
  }
  mx = red[0]; __syncthreads();
  float sm = 0.f;
  for (int j = tid; j < M; j += T) { float e = expf(p[j] - mx); p[j] = e; sm += e; }
  red[tid] = sm; __syncthreads();
  for (int off = T >> 1; off; off >>= 1) {
    if (tid < off) red[tid] += red[tid + off];
    __syncthreads();
  }
  float inv = 1.f / red[0];
  for (int j = tid; j < M; j += T) p[j] *= inv;
}

__global__ void k_colsum(const float* __restrict__ att, int ldm, long satt,
                         float* __restrict__ cols, int Nrows, int M, int B)
{
  long t = (long)blockIdx.x * blockDim.x + threadIdx.x;
  if (t >= (long)B * M) return;
  int b = (int)(t / M), j = (int)(t % M);
  const float* p = att + (long)b * satt + j;
  float s = 0.f;
  for (int i = 0; i < Nrows; ++i) s += p[(long)i * ldm];
  cols[(long)b * M + j] = s;
}

__global__ void k_rescale(float* __restrict__ att, int ldm, long satt,
                          const float* __restrict__ cols, int M, int Nrows, int B)
{
  long t = (long)blockIdx.x * blockDim.x + threadIdx.x;
  if (t >= (long)B * Nrows * M) return;
  int j = (int)(t % M);
  long r = t / M;
  int i = (int)(r % Nrows), b = (int)(r / Nrows);
  att[(long)b * satt + (long)i * ldm + j] /= (1e-9f + cols[(long)b * M + j]);
}

// ---------------------------------------------------------------------------
// kNN over a precomputed Gram matrix (feature space): top-16 of
// neg_d = 2*G[n,m] - |f_n|^2 - |f_m|^2  (first-index-wins on ties)
// ---------------------------------------------------------------------------
__global__ void k_knn_gram(const float* __restrict__ G, const float* __restrict__ sq,
                           int* __restrict__ idx, int N)
{
  int n = blockIdx.x * blockDim.x + threadIdx.x;
  if (n >= N) return;
  float val[16]; int id[16];
  for (int j = 0; j < 16; ++j) { val[j] = NEG_BIG; id[j] = 0; }
  float sn = sq[n];
  const float* g = G + (long)n * N;
  for (int m = 0; m < N; ++m) {
    float nd = 2.f * g[m] - sn - sq[m];
    if (nd > val[15]) {
      int p = 15;
      while (p > 0 && val[p - 1] < nd) { val[p] = val[p - 1]; id[p] = id[p - 1]; --p; }
      val[p] = nd; id[p] = m;
    }
  }
  for (int j = 0; j < 16; ++j) idx[(long)n * 16 + j] = id[j];
}

// Fused DenseGCN block: per point, loop k=16 neighbors through
// conv(first)->relu, [y1,x], conv(stack1)->relu, [y1,x,y2], conv(stack2),
// channel order [y1(24), x(C), y2(24), y3(24)], max over k, outer ReLU.
__global__ void k_dense(const float* __restrict__ x, int xld, long xs,
                        int C, int N, const int* __restrict__ idx,
                        const float* __restrict__ wf,  const float* __restrict__ bf,
                        const float* __restrict__ ws1, const float* __restrict__ bs1,
                        const float* __restrict__ ws2, const float* __restrict__ bs2,
                        float* __restrict__ out, int oln, long os, int B)
{
  long t = (long)blockIdx.x * blockDim.x + threadIdx.x;
  if (t >= (long)B * N) return;
  int b = (int)(t / N), n = (int)(t % N);
  const float* xb = x + (long)b * xs;
  float ctr[48];
  for (int c = 0; c < C; ++c) ctr[c] = xb[(long)c * xld + n];
  const int Ct = C + 72;
  float best[120];
  for (int c = 0; c < Ct; ++c) best[c] = NEG_BIG;
  const int c2 = 2 * C, in1 = 24 + C, in2 = 48 + C;
  for (int j = 0; j < 16; ++j) {
    int m = idx[((long)b * N + n) * 16 + j];
    float nb[48];
    for (int c = 0; c < C; ++c) nb[c] = xb[(long)c * xld + m];
    float y1[24];
    for (int o = 0; o < 24; ++o) {
      float a = bf[o];
      const float* w = wf + (long)o * c2;
      for (int c = 0; c < C; ++c) a += w[c] * (nb[c] - ctr[c]);
      for (int c = 0; c < C; ++c) a += w[C + c] * ctr[c];
      y1[o] = fmaxf(a, 0.f);
    }
    float y2[24];
    for (int o = 0; o < 24; ++o) {
      float a = bs1[o];
      const float* w = ws1 + (long)o * in1;
      for (int c = 0; c < 24; ++c) a += w[c] * y1[c];
      for (int c = 0; c < C; ++c) a += w[24 + c] * ctr[c];
      y2[o] = fmaxf(a, 0.f);
    }
    float y3[24];
    for (int o = 0; o < 24; ++o) {
      float a = bs2[o];
      const float* w = ws2 + (long)o * in2;
      for (int c = 0; c < 24; ++c) a += w[c] * y1[c];
      for (int c = 0; c < C; ++c) a += w[24 + c] * ctr[c];
      for (int c = 0; c < 24; ++c) a += w[24 + C + c] * y2[c];
      y3[o] = a;
    }
    for (int o = 0; o < 24; ++o) best[o]          = fmaxf(best[o],          y1[o]);
    for (int c = 0; c < C;  ++c) best[24 + c]     = fmaxf(best[24 + c],     ctr[c]);
    for (int o = 0; o < 24; ++o) best[24 + C + o] = fmaxf(best[24 + C + o], y2[o]);
    for (int o = 0; o < 24; ++o) best[48 + C + o] = fmaxf(best[48 + C + o], y3[o]);
  }
  float* ob = out + (long)b * os;
  for (int c = 0; c < Ct; ++c)
    ob[(long)c * oln + n] = fmaxf(best[c], 0.f);   // outer relu
}

// ---------------------------------------------------------------------------
// Farthest-point sampling: one block per batch, LDS distance array,
// first-max ties like jnp.argmax.
// ---------------------------------------------------------------------------
__global__ void k_fps(const float* __restrict__ pts, int* __restrict__ oidx,
                      int N, int S)
{
  __shared__ float dist[3072];
  __shared__ float rv[256];
  __shared__ int   ri[256];
  int b = blockIdx.x, tid = threadIdx.x, T = blockDim.x;
  const float* pb = pts + (long)b * N * 3;
  for (int i = tid; i < N; i += T) dist[i] = 1e10f;
  if (tid == 0) oidx[(long)b * S] = 0;
  __syncthreads();
  int last = 0;
  for (int s = 1; s < S; ++s) {
    float lx = pb[last * 3], ly = pb[last * 3 + 1], lz = pb[last * 3 + 2];
    float bv = NEG_BIG; int bi = 0x7fffffff;
    for (int i = tid; i < N; i += T) {
      float dx = pb[i * 3] - lx, dy = pb[i * 3 + 1] - ly, dz = pb[i * 3 + 2] - lz;
      float d = dx * dx + dy * dy + dz * dz;
      float nd = fminf(dist[i], d);
      dist[i] = nd;
      if (nd > bv || (nd == bv && i < bi)) { bv = nd; bi = i; }
    }
    rv[tid] = bv; ri[tid] = bi; __syncthreads();
    for (int off = T >> 1; off; off >>= 1) {
      if (tid < off) {
        if (rv[tid + off] > rv[tid] ||
            (rv[tid + off] == rv[tid] && ri[tid + off] < ri[tid])) {
          rv[tid] = rv[tid + off]; ri[tid] = ri[tid + off];
        }
      }
      __syncthreads();
    }
    last = ri[0];
    if (tid == 0) oidx[(long)b * S + s] = last;
    __syncthreads();
  }
}

__global__ void k_transpose_pts(const float* __restrict__ x, float* __restrict__ pc,
                                int N, int B)
{
  long t = (long)blockIdx.x * blockDim.x + threadIdx.x;
  if (t >= (long)B * N) return;
  int b = (int)(t / N), n = (int)(t % N);
  for (int i = 0; i < 3; ++i)
    pc[((long)b * N + n) * 3 + i] = x[((long)b * 3 + i) * N + n];
}

__global__ void k_gather_pts(float* __restrict__ dst, const float* __restrict__ src,
                             const int* __restrict__ pidx, int S, int N, int B)
{
  long t = (long)blockIdx.x * blockDim.x + threadIdx.x;
  if (t >= (long)B * S) return;
  int b = (int)(t / S), s = (int)(t % S);
  int pi = pidx[(long)b * S + s];
  for (int i = 0; i < 3; ++i)
    dst[((long)b * S + s) * 3 + i] = src[((long)b * N + pi) * 3 + i];
}

__global__ void k_knn_pts(const float* __restrict__ npnts, const float* __restrict__ pts,
                          int* __restrict__ idx, int N, int S, int B)
{
  long t = (long)blockIdx.x * blockDim.x + threadIdx.x;
  if (t >= (long)B * S) return;
  int b = (int)(t / S), s = (int)(t % S);
  const float* q  = npnts + ((long)b * S + s) * 3;
  const float* pb = pts + (long)b * N * 3;
  float qx = q[0], qy = q[1], qz = q[2];
  float val[16]; int id[16];
  for (int j = 0; j < 16; ++j) { val[j] = 3.0e38f; id[j] = 0; }
  for (int m = 0; m < N; ++m) {
    float dx = pb[m * 3] - qx, dy = pb[m * 3 + 1] - qy, dz = pb[m * 3 + 2] - qz;
    float d = dx * dx + dy * dy + dz * dz;
    if (d < val[15]) {
      int p = 15;
      while (p > 0 && val[p - 1] > d) { val[p] = val[p - 1]; id[p] = id[p - 1]; --p; }
      val[p] = d; id[p] = m;
    }
  }
  for (int j = 0; j < 16; ++j) idx[((long)b * S + s) * 16 + j] = id[j];
}

// Edge-preserve sampling assembly: out[b, 0:C] = feat[:, p_idx],
// out[b, C:2C] = max over k of feat[:, pn_idx].
__global__ void k_eps_build(const float* __restrict__ f, int fld, long fs, int C,
                            const int* __restrict__ pidx, const int* __restrict__ pnidx,
                            int S, float* __restrict__ out, int B)
{
  long t = (long)blockIdx.x * blockDim.x + threadIdx.x;
  long tot = (long)B * 2 * C * S;
  if (t >= tot) return;
  int s = (int)(t % S);
  long r = t / S;
  int c2 = (int)(r % (2 * C)), b = (int)(r / (2 * C));
  const float* fb = f + (long)b * fs;
  float v;
  if (c2 < C) {
    v = fb[(long)c2 * fld + pidx[(long)b * S + s]];
  } else {
    int c = c2 - C;
    v = NEG_BIG;
    for (int j = 0; j < 16; ++j) {
      int m = pnidx[((long)b * S + s) * 16 + j];
      v = fmaxf(v, fb[(long)c * fld + m]);
    }
  }
  out[((long)b * 2 * C + c2) * S + s] = v;
}

__global__ void k_rowmax(const float* __restrict__ gm, float* __restrict__ gfv, int Nn)
{
  long t = (long)blockIdx.x * blockDim.x + threadIdx.x;
  if (t >= 4L * 1024) return;
  const float* p = gm + t * Nn;
  float v = NEG_BIG;
  for (int n = 0; n < Nn; ++n) v = fmaxf(v, p[n]);
  gfv[t] = v;
}

__global__ void k_bcast_gf(float* __restrict__ c5, const float* __restrict__ t2, int B)
{
  long t = (long)blockIdx.x * blockDim.x + threadIdx.x;
  if (t >= (long)B * 1024 * 64) return;
  int n = (int)(t % 64);
  long r = t / 64;
  int c = (int)(r % 1024), b = (int)(r / 1024);
  c5[((long)b * 2824 + c) * 64 + n] = t2[c * 4 + b];
}

// ---------------------------------------------------------------------------
// Host orchestration
// ---------------------------------------------------------------------------
static inline void gemm(hipStream_t st, const float* A, const float* B, float* C,
                        const float* bias, int M, int N, int K,
                        int lda, int ldb, int ldc, long sA, long sB, long sC,
                        int batch, bool tA, bool tB, bool relu,
                        const float* rowv = nullptr, const float* colv = nullptr,
                        const float* rb = nullptr, long sR = 0, long sCl = 0)
{
  dim3 g(cdiv(N, 32), cdiv(M, 16), batch);
  int r = relu ? 1 : 0;
  if (!tA && !tB)
    k_gemm_t<0,0><<<g, 32, 0, st>>>(A, B, C, bias, M, N, K, lda, ldb, ldc,
                                    sA, sB, sC, r, rowv, colv, rb, sR, sCl);
  else if (tA && !tB)
    k_gemm_t<1,0><<<g, 32, 0, st>>>(A, B, C, bias, M, N, K, lda, ldb, ldc,
                                    sA, sB, sC, r, rowv, colv, rb, sR, sCl);
  else if (!tA && tB)
    k_gemm_t<0,1><<<g, 32, 0, st>>>(A, B, C, bias, M, N, K, lda, ldb, ldc,
                                    sA, sB, sC, r, rowv, colv, rb, sR, sCl);
  else
    k_gemm_t<1,1><<<g, 32, 0, st>>>(A, B, C, bias, M, N, K, lda, ldb, ldc,
                                    sA, sB, sC, r, rowv, colv, rb, sR, sCl);
}

extern "C" void kernel_launch(void* const* d_in, const int* in_sizes, int n_in,
                              void* d_out, int out_size, void* d_ws, size_t ws_size,
                              hipStream_t stream)
{
  (void)in_sizes; (void)n_in; (void)out_size; (void)ws_size;
  const int B = 4, NC = 1024, NP = 2048, NA = 3072;
  const int NT = 256;

  const float* x = (const float*)d_in[0];
  auto F = [&](int i) { return (const float*)d_in[i]; };
  // params flattened in jax sorted-key order
  const float *conv1_b=F(1),*conv1_w=F(2),*conv1p_b=F(3),*conv1p_w=F(4);
  const float *conv2_b=F(5),*conv2_w=F(6),*conv3_b=F(7),*conv3_w=F(8);
  const float *conv4_b=F(9),*conv4_w=F(10),*conv5_b=F(11),*conv5_w=F(12);
  const float *catt_b=F(13),*catt_w=F(14);
  const float *dc1_fb=F(15),*dc1_fw=F(16),*dc1_1b=F(17),*dc1_1w=F(18),*dc1_2b=F(19),*dc1_2w=F(20);
  const float *dcp_fb=F(21),*dcp_fw=F(22),*dcp_1b=F(23),*dcp_1w=F(24),*dcp_2b=F(25),*dcp_2w=F(26);
  const float *dc2_fb=F(27),*dc2_fw=F(28),*dc2_1b=F(29),*dc2_1w=F(30),*dc2_2b=F(31),*dc2_2w=F(32);
  const float *dc3_fb=F(33),*dc3_fw=F(34),*dc3_1b=F(35),*dc3_1w=F(36),*dc3_2b=F(37),*dc3_2w=F(38);
  const float *dc4_fb=F(39),*dc4_fw=F(40),*dc4_1b=F(41),*dc4_1w=F(42),*dc4_2b=F(43),*dc4_2w=F(44);
  const float *fc1_b=F(45),*fc1_w=F(46),*fc2_b=F(47),*fc2_w=F(48);
  const float *gf_b=F(49),*gf_w=F(50);
  const float *sa_beta=F(51),*sa_gamma=F(52),*sa_posb=F(53),*sa_posw=F(54);
  const float *sa_tb=F(55),*sa_tw=F(56),*sa_vb=F(57),*sa_vw=F(58),*sa_wqk=F(59);
  const float *g_beta=F(60),*g_gamma=F(61),*g_tb=F(62),*g_tw=F(63);
  const float *g_vb=F(64),*g_vw=F(65),*g_wqk=F(66);

  // workspace allocator (float buffers, 256B aligned)
  char* wp = (char*)d_ws;
  size_t used = 0;
  auto allocF = [&](size_t e)->float*{ float* p=(float*)(wp+used); used += ((e*4+255)/256)*256; return p; };
  auto allocI = [&](size_t e)->int*  { int*   p=(int*)  (wp+used); used += ((e*4+255)/256)*256; return p; };

  float* PE   = allocF((size_t)B*21*NA);
  float* X0   = allocF((size_t)B*24*NA);
  float* XC1  = allocF((size_t)B*120*NA);
  float* PALL = allocF((size_t)B*30*NA);
  float* GR   = allocF((size_t)NA*NA);            // shared scratch (gram / attention)
  float* PROJ = allocF((size_t)B*NA);
  float* COLS = allocF((size_t)B*NA);
  float* SQN  = allocF((size_t)B*NA);
  float* XV   = allocF((size_t)B*120*NA);
  float* XR   = allocF((size_t)B*120*NA);
  float* TB   = allocF((size_t)B*120*NA);
  float* TT   = allocF((size_t)B*120*NA);
  float* X1G  = allocF((size_t)B*120*NA);
  float* X1F  = allocF((size_t)B*120*NA);
  float* PC1  = allocF((size_t)B*NA*3);
  float* PC2  = allocF((size_t)B*1024*3);
  float* PC3  = allocF((size_t)B*256*3);
  float* PC4  = allocF((size_t)B*64*3);
  float* X1D  = allocF((size_t)B*240*1024);
  float* X2A  = allocF((size_t)B*48*1024);
  float* X2C  = allocF((size_t)B*360*1024);
  float* X2D  = allocF((size_t)B*720*256);
  float* X3A  = allocF((size_t)B*48*256);
  float* X3C  = allocF((size_t)B*840*256);
  float* X3D  = allocF((size_t)B*1680*64);
  float* X4A  = allocF((size_t)B*48*64);
  float* X4C  = allocF((size_t)B*1800*64);
  float* GFM  = allocF((size_t)B*1024*64);
  float* GFV  = allocF((size_t)B*1024);
  float* T1   = allocF((size_t)512*4);
  float* T2   = allocF((size_t)1024*4);
  float* C5IN = allocF((size_t)B*2824*64);
  int*   IDX  = allocI((size_t)B*NA*16);
  int*   PIDX = allocI((size_t)B*1024);
  int*   PNIX = allocI((size_t)B*1024*16);

  auto EW = [&](long tot){ return dim3(cdiv(tot, NT)); };

  // ---- positional encoding over all 3072 columns (pointwise) ----
  k_pos_enc<<<EW((long)B*NA), NT, 0, stream>>>(x, PE, NA, B);

  // ---- conv1 / conv1p -> X0 (24 ch, cols [0,1024)=coarse, [1024,3072)=partial)
  gemm(stream, conv1_w,  PE,        X0,        conv1_b,  24, NC, 21, 21, NA, NA,
       0, (long)21*NA, (long)24*NA, B, false, false, true);
  gemm(stream, conv1p_w, PE + NC,   X0 + NC,   conv1p_b, 24, NP, 21, 21, NA, NA,
       0, (long)21*NA, (long)24*NA, B, false, false, true);

  // ---- DenseGCN helper: gram (WMMA) -> top-16 -> fused dense conv ----
  auto dense_block = [&](const float* xin, int xld, long xs, int C, int Nn,
                         const float* fw, const float* fb, const float* s1w,
                         const float* s1b, const float* s2w, const float* s2b,
                         float* outBase, int oln, long os){
    k_sqnorm<<<EW((long)B*Nn), NT, 0, stream>>>(xin, xld, xs, SQN, C, Nn, B);
    for (int b = 0; b < B; ++b) {
      gemm(stream, xin + (long)b*xs, xin + (long)b*xs, GR, nullptr,
           Nn, Nn, C, xld, xld, Nn, 0, 0, 0, 1, true, false, false);
      k_knn_gram<<<cdiv(Nn, NT), NT, 0, stream>>>(GR, SQN + (long)b*Nn,
                                                  IDX + (long)b*Nn*16, Nn);
    }
    k_dense<<<EW((long)B*Nn), NT, 0, stream>>>(xin, xld, xs, C, Nn, IDX,
        fw, fb, s1w, s1b, s2w, s2b, outBase, oln, os, B);
  };

  // dc1 (coarse) -> XC1 ch[0..96), then x0 -> ch[96..120)
  dense_block(X0, NA, (long)24*NA, 24, NC, dc1_fw, dc1_fb, dc1_1w, dc1_1b,
              dc1_2w, dc1_2b, XC1, NA, (long)120*NA);
  k_copy<<<EW((long)B*24*NC), NT, 0, stream>>>(XC1 + (long)96*NA, NA, (long)120*NA,
                                               X0, NA, (long)24*NA, 24, NC, B);
  // dc1p (partial) -> XC1 cols [1024,3072)
  dense_block(X0 + NC, NA, (long)24*NA, 24, NP, dcp_fw, dcp_fb, dcp_1w, dcp_1b,
              dcp_2w, dcp_2b, XC1 + NC, NA, (long)120*NA);
  k_copy<<<EW((long)B*24*NP), NT, 0, stream>>>(XC1 + (long)96*NA + NC, NA, (long)120*NA,
                                               X0 + NC, NA, (long)24*NA, 24, NP, B);

  // ---- cross-feature SA (queries=coarse 1024, keys=partial 2048) ----
  gemm(stream, sa_wqk, XC1, PALL, nullptr, 30, NA, 120, 120, NA, NA,
       0, (long)120*NA, (long)30*NA, B, false, false, false);
  k_proj<<<EW((long)B*NA), NT, 0, stream>>>(PE, sa_posw, PROJ, NA, B);
  // energy = q^T k + (proj_q - proj_k + pos_b), batched into GR (B,1024,2048)
  gemm(stream, PALL, PALL + NC, GR, nullptr, NC, NP, 30, NA, NA, NP,
       (long)30*NA, (long)30*NA, (long)NC*NP, B, true, false, false,
       PROJ, PROJ + NC, sa_posb, NA, NA);
  k_softmax_rows<<<dim3(NC, B), NT, 0, stream>>>(GR, NP, (long)NC*NP, NP);
  k_colsum<<<EW((long)B*NP), NT, 0, stream>>>(GR, NP, (long)NC*NP, COLS, NC, NP, B);
  k_rescale<<<EW((long)B*NC*NP), NT, 0, stream>>>(GR, NP, (long)NC*NP, COLS, NP, NC, B);
  // xv = v-conv on partial; xr = xv * att^T
  gemm(stream, sa_vw, XC1 + NC, XV, sa_vb, 120, NP, 120, 120, NA, NP,
       0, (long)120*NA, (long)120*NP, B, false, false, false);
  gemm(stream, XV, GR, XR, nullptr, 120, NC, NP, NP, NP, NC,
       (long)120*NP, (long)NC*NP, (long)120*NC, B, false, true, false);
  k_sub<<<EW((long)B*120*NC), NT, 0, stream>>>(TB, NC, (long)120*NC,
       XC1, NA, (long)120*NA, XR, NC, (long)120*NC, 120, NC, B);
  gemm(stream, sa_tw, TB, TT, sa_tb, 120, NC, 120, 120, NC, NC,
       0, (long)120*NC, (long)120*NC, B, false, false, false);
  k_bn_res<<<EW((long)B*120*NC), NT, 0, stream>>>(X1G, NA, (long)120*NA,
       XC1, NA, (long)120*NA, TT, NC, (long)120*NC, sa_gamma, sa_beta, 120, NC, B);
  // conv_att on partial part -> X1G cols [1024,3072)
  gemm(stream, catt_w, XC1 + NC, X1G + NC, catt_b, 120, NP, 120, 120, NA, NA,
       0, (long)120*NA, (long)120*NA, B, false, false, true);

  // ---- global SA on (B,120,3072); per-batch 3072x3072 attention in GR ----
  gemm(stream, g_wqk, X1G, PALL, nullptr, 30, NA, 120, 120, NA, NA,
       0, (long)120*NA, (long)30*NA, B, false, false, false);
  gemm(stream, g_vw, X1G, XV, g_vb, 120, NA, 120, 120, NA, NA,
       0, (long)120*NA, (long)120*NA, B, false, false, false);
  for (int b = 0; b < B; ++b) {
    const float* Pb = PALL + (long)b*30*NA;
    gemm(stream, Pb, Pb, GR, nullptr, NA, NA, 30, NA, NA, NA,
         0, 0, 0, 1, true, false, false);
    k_softmax_rows<<<dim3(NA, 1), NT, 0, stream>>>(GR, NA, 0, NA);
    k_colsum<<<EW((long)NA), NT, 0, stream>>>(GR, NA, 0, COLS, NA, NA, 1);
    k_rescale<<<EW((long)NA*NA), NT, 0, stream>>>(GR, NA, 0, COLS, NA, NA, 1);
    gemm(stream, XV + (long)b*120*NA, GR, XR + (long)b*120*NA, nullptr,
         120, NA, NA, NA, NA, NA, 0, 0, 0, 1, false, true, false);
  }
  k_sub<<<EW((long)B*120*NA), NT, 0, stream>>>(TB, NA, (long)120*NA,
       X1G, NA, (long)120*NA, XR, NA, (long)120*NA, 120, NA, B);
  gemm(stream, g_tw, TB, TT, g_tb, 120, NA, 120, 120, NA, NA,
       0, (long)120*NA, (long)120*NA, B, false, false, false);
  k_bn_res<<<EW((long)B*120*NA), NT, 0, stream>>>(X1F, NA, (long)120*NA,
       X1G, NA, (long)120*NA, TT, NA, (long)120*NA, g_gamma, g_beta, 120, NA, B);

  // ---- edge-preserve sampling helper ----
  auto eps = [&](const float* feat, int fld, long fs, int C,
                 const float* psrc, int Nn, int S, float* pdst, float* outBuf){
    k_fps<<<B, 256, 0, stream>>>(psrc, PIDX, Nn, S);
    k_gather_pts<<<EW((long)B*S), NT, 0, stream>>>(pdst, psrc, PIDX, S, Nn, B);
    k_knn_pts<<<EW((long)B*S), NT, 0, stream>>>(pdst, psrc, PNIX, Nn, S, B);
    k_eps_build<<<EW((long)B*2*C*S), NT, 0, stream>>>(feat, fld, fs, C, PIDX, PNIX,
                                                      S, outBuf, B);
  };

  // level 1: 3072 -> 1024
  k_transpose_pts<<<EW((long)B*NA), NT, 0, stream>>>(x, PC1, NA, B);
  eps(X1F, NA, (long)120*NA, 120, PC1, NA, 1024, PC2, X1D);
  gemm(stream, conv2_w, X1D, X2A, conv2_b, 48, 1024, 240, 240, 1024, 1024,
       0, (long)240*1024, (long)48*1024, B, false, false, true);
  dense_block(X2A, 1024, (long)48*1024, 48, 1024, dc2_fw, dc2_fb, dc2_1w, dc2_1b,
              dc2_2w, dc2_2b, X2C, 1024, (long)360*1024);
  k_copy<<<EW((long)B*240*1024), NT, 0, stream>>>(X2C + (long)120*1024, 1024,
       (long)360*1024, X1D, 1024, (long)240*1024, 240, 1024, B);

  // level 2: 1024 -> 256
  eps(X2C, 1024, (long)360*1024, 360, PC2, 1024, 256, PC3, X2D);
  gemm(stream, conv3_w, X2D, X3A, conv3_b, 48, 256, 720, 720, 256, 256,
       0, (long)720*256, (long)48*256, B, false, false, true);
  dense_block(X3A, 256, (long)48*256, 48, 256, dc3_fw, dc3_fb, dc3_1w, dc3_1b,
              dc3_2w, dc3_2b, X3C, 256, (long)840*256);
  k_copy<<<EW((long)B*720*256), NT, 0, stream>>>(X3C + (long)120*256, 256,
       (long)840*256, X2D, 256, (long)720*256, 720, 256, B);

  // level 3: 256 -> 64
  eps(X3C, 256, (long)840*256, 840, PC3, 256, 64, PC4, X3D);
  gemm(stream, conv4_w, X3D, X4A, conv4_b, 48, 64, 1680, 1680, 64, 64,
       0, (long)1680*64, (long)48*64, B, false, false, true);
  dense_block(X4A, 64, (long)48*64, 48, 64, dc4_fw, dc4_fb, dc4_1w, dc4_1b,
              dc4_2w, dc4_2b, X4C, 64, (long)1800*64);
  k_copy<<<EW((long)B*1680*64), NT, 0, stream>>>(X4C + (long)120*64, 64,
       (long)1800*64, X3D, 64, (long)1680*64, 1680, 64, B);

  // ---- global feature head ----
  gemm(stream, gf_w, X4C, GFM, gf_b, 1024, 64, 1800, 1800, 64, 64,
       0, (long)1800*64, (long)1024*64, B, false, false, false);
  k_rowmax<<<EW((long)B*1024), NT, 0, stream>>>(GFM, GFV, 64);
  gemm(stream, fc1_w, GFV, T1, fc1_b, 512, 4, 1024, 1024, 1024, 4,
       0, 0, 0, 1, false, true, true);
  gemm(stream, fc2_w, T1, T2, fc2_b, 1024, 4, 512, 512, 4, 4,
       0, 0, 0, 1, false, false, true);
  k_bcast_gf<<<EW((long)B*1024*64), NT, 0, stream>>>(C5IN, T2, B);
  k_copy<<<EW((long)B*1800*64), NT, 0, stream>>>(C5IN + (long)1024*64, 64,
       (long)2824*64, X4C, 64, (long)1800*64, 1800, 64, B);

  // ---- conv5 -> output (B,1024,64) ----
  gemm(stream, conv5_w, C5IN, (float*)d_out, conv5_b, 1024, 64, 2824,
       2824, 64, 64, 0, (long)2824*64, (long)1024*64, B, false, false, true);
}
